// GAT_64338610094368
// MI455X (gfx1250) — matrix operand order repaired and checked by reading, not compiled, not run on hardware
//
#include <hip/hip_runtime.h>
#include <hip/hip_bf16.h>
#include <math.h>

// GAT over word/doc graph for MI455X (gfx1250, wave32, WMMA).
// GEMMs use v_wmma_f32_16x16x32_bf16 with bf16 operands pre-converted in
// global memory (activations row-major, weights transposed), staged to LDS
// with GLOBAL_LOAD_ASYNC_TO_LDS_B128 (ASYNCcnt) double-buffered so the
// prefetch of tile k+1 overlaps the WMMAs of tile k. Fragment-ready LDS
// layouts: each WMMA fragment is two aligned ds_load_b128. Edge softmax via
// segment atomics; edge-embedding GEMM collapsed to a 10x8 score table.

typedef __bf16 bf16_t;
typedef __attribute__((ext_vector_type(8)))  __bf16 v8bf;
typedef __attribute__((ext_vector_type(16))) __bf16 v16bf;
typedef __attribute__((ext_vector_type(8)))  float  v8f;
typedef __attribute__((ext_vector_type(4)))  float  f32x4;

#define NHEAD 8
#define HDIM  64
#define DM    512
#define FFNH  2048
#define FFN_CHUNK 2048

#ifndef USE_ASYNC_LDS
#define USE_ASYNC_LDS 1   // flip to 0 to fall back to sync LDS staging
#endif

static inline int cdiv(int a, int b) { return (a + b - 1) / b; }

// ---------------------------------------------------------------------------
// Async 16-byte global->LDS copy (ASYNCcnt). LDS offset = low 32 bits of the
// generic pointer (ISA 10.2 aperture).
// ---------------------------------------------------------------------------
__device__ __forceinline__ void async_ld16(unsigned lds_off, const bf16_t* g) {
#if USE_ASYNC_LDS
  asm volatile("global_load_async_to_lds_b128 %0, %1, off"
               :: "v"(lds_off), "v"(g) : "memory");
#else
  (void)lds_off; (void)g;
#endif
}
__device__ __forceinline__ void async_wait0() {
#if USE_ASYNC_LDS
  asm volatile("s_wait_asynccnt 0x0" ::: "memory");
#endif
}

// ---------------------------------------------------------------------------
// GEMM: C[M,N] = act(A[M,K] @ B[K,N] + bias), C f32.
// A: bf16 row-major [M][K]. Bt: bf16 TRANSPOSED [N][K]. act: 0=bias, 1=relu.
// Requires N % 64 == 0, K % 64 == 0 (>=2 K-steps). M guarded (row clamp on
// loads, masked stores). Block = 256 threads = 8 wave32; tile 128(M) x 64(N);
// K-step 32; double-buffered async LDS staging.
// Wave w owns rows [w*16,w*16+16) x all four 16-col subtiles: 4 WMMA/K-step.
//
// LDS (40-element pitch = 80 B; 16 lanes hit disjoint bank quads):
//  As[b][row][k]: chunks in source K order (K0-7,K8-15,K16-23,K24-31) so a lane's
//    A fragment (elem t = K hi*16+t) is 32 B at
//    [row][hi*16].
//  Bs[b][col][k]: a lane's B fragment (elem t = K hi*16+t) is 32 B at
//    [col][hi*16].
// ---------------------------------------------------------------------------
__global__ __launch_bounds__(256) void gat_gemm_bf16(
    const bf16_t* __restrict__ A, const bf16_t* __restrict__ Bt,
    const float* __restrict__ bias, float* __restrict__ C,
    int M, int N, int K, int act)
{
  __shared__ bf16_t As[2][128][40];
  __shared__ bf16_t Bs[2][64][40];

  const int tid  = threadIdx.x;
  const int lane = tid & 31;
  const int wave = tid >> 5;
  const int bm = blockIdx.x * 128;
  const int bn = blockIdx.y * 64;
  const int hi = lane >> 4;
  const int lo = lane & 15;

  // A staging: thread covers rows {ar, ar+64}, 8 K-values of chunk aq
  const int ar   = tid >> 2;                              // 0..63
  const int aq   = tid & 3;                               // K chunk
  const int asrc = aq * 8;                                // src K: 0,8,16,24
  // B staging: thread covers column bc, K range [bk, bk+8)
  const int bc = tid & 63;
  const int bk = (tid >> 6) * 8;

  // clamped global rows (out-of-range results are masked in the epilogue)
  int gr0 = bm + ar;        if (gr0 > M - 1) gr0 = M - 1;
  int gr1 = bm + ar + 64;   if (gr1 > M - 1) gr1 = M - 1;
  const bf16_t* a0p = A + (size_t)gr0 * K + asrc;
  const bf16_t* a1p = A + (size_t)gr1 * K + asrc;
  const bf16_t* btp = Bt + (size_t)(bn + bc) * K + bk;

  unsigned lA0[2], lA1[2], lB[2];
#pragma unroll
  for (int b = 0; b < 2; ++b) {
    lA0[b] = (unsigned)(size_t)&As[b][ar][aq * 8];
    lA1[b] = (unsigned)(size_t)&As[b][ar + 64][aq * 8];
    lB[b]  = (unsigned)(size_t)&Bs[b][bc][bk];
  }

  v8f acc[4] = {{}, {}, {}, {}};

  // prologue: stage K-tile 0 into buffer 0
#if USE_ASYNC_LDS
  async_ld16(lA0[0], a0p);
  async_ld16(lA1[0], a1p);
  async_ld16(lB[0],  btp);
  async_wait0();
#else
  *(v8bf*)&As[0][ar][aq * 8]      = *(const v8bf*)a0p;
  *(v8bf*)&As[0][ar + 64][aq * 8] = *(const v8bf*)a1p;
  *(v8bf*)&Bs[0][bc][bk]          = *(const v8bf*)btp;
#endif
  __syncthreads();

  int cur = 0;
  for (int k0 = 0; k0 < K; k0 += 32) {
    const int nxt = cur ^ 1;
    // prefetch next K-tile into the other buffer (overlaps the WMMAs below)
    if (k0 + 32 < K) {
#if USE_ASYNC_LDS
      async_ld16(lA0[nxt], a0p + k0 + 32);
      async_ld16(lA1[nxt], a1p + k0 + 32);
      async_ld16(lB[nxt],  btp + k0 + 32);
#else
      *(v8bf*)&As[nxt][ar][aq * 8]      = *(const v8bf*)(a0p + k0 + 32);
      *(v8bf*)&As[nxt][ar + 64][aq * 8] = *(const v8bf*)(a1p + k0 + 32);
      *(v8bf*)&Bs[nxt][bc][bk]          = *(const v8bf*)(btp + k0 + 32);
#endif
    }

    const int arow = wave * 16 + lo;
    v8bf a0 = *(const v8bf*)&As[cur][arow][hi * 16];
    v8bf a1 = *(const v8bf*)&As[cur][arow][hi * 16 + 8];
    v16bf af = __builtin_shufflevector(a0, a1,
        0, 1, 2, 3, 4, 5, 6, 7, 8, 9, 10, 11, 12, 13, 14, 15);

    v16bf bfr[4];
#pragma unroll
    for (int j = 0; j < 4; ++j) {
      v8bf b0 = *(const v8bf*)&Bs[cur][j * 16 + lo][hi * 16];
      v8bf b1 = *(const v8bf*)&Bs[cur][j * 16 + lo][hi * 16 + 8];
      bfr[j] = __builtin_shufflevector(b0, b1,
          0, 1, 2, 3, 4, 5, 6, 7, 8, 9, 10, 11, 12, 13, 14, 15);
    }
#pragma unroll
    for (int j = 0; j < 4; ++j)
      acc[j] = __builtin_amdgcn_wmma_f32_16x16x32_bf16(
          false, af, false, bfr[j], (short)0, acc[j], false, false);

    async_wait0();
    __syncthreads();
    cur = nxt;
  }

  // epilogue: C/D layout: VGPR r = row r (lanes 0-15) / r+8 (lanes 16-31)
#pragma unroll
  for (int j = 0; j < 4; ++j) {
    int colj = bn + j * 16 + lo;
    float bb = bias ? bias[colj] : 0.0f;
#pragma unroll
    for (int r = 0; r < 8; ++r) {
      int row = bm + wave * 16 + hi * 8 + r;
      if (row < M) {
        float v = acc[j][r] + bb;
        if (act == 1) v = fmaxf(v, 0.0f);
        C[(size_t)row * N + colj] = v;
      }
    }
  }
}

// ---------------------------------------------------------------------------
// Conversion kernels
// ---------------------------------------------------------------------------
// out[i] = bf16(in[i]); n % 8 == 0
__global__ void gat_cvt_bf16(const float* __restrict__ in,
                             bf16_t* __restrict__ out, long long n) {
  long long i = ((long long)blockIdx.x * blockDim.x + threadIdx.x) * 8;
  long long st = (long long)gridDim.x * blockDim.x * 8;
  for (; i < n; i += st) {
    f32x4 u0 = *(const f32x4*)(in + i);
    f32x4 u1 = *(const f32x4*)(in + i + 4);
    v8bf p;
#pragma unroll
    for (int t = 0; t < 4; ++t) {
      p[t]     = (bf16_t)u0[t];
      p[t + 4] = (bf16_t)u1[t];
    }
    *(v8bf*)(out + i) = p;
  }
}

// out[n][k] = bf16(in[k][n]); in: K x N f32; K, N % 32 == 0. LDS-tiled.
__global__ __launch_bounds__(256) void gat_cvt_t_bf16(
    const float* __restrict__ in, bf16_t* __restrict__ out, int K, int N) {
  __shared__ float tile[32][33];
  int kb = blockIdx.x * 32, nb = blockIdx.y * 32;
  int tx = threadIdx.x & 31, ty = threadIdx.x >> 5;  // ty: 0..7
#pragma unroll
  for (int i = 0; i < 32; i += 8)
    tile[ty + i][tx] = in[(size_t)(kb + ty + i) * N + (nb + tx)];
  __syncthreads();
#pragma unroll
  for (int i = 0; i < 32; i += 8)
    out[(size_t)(nb + ty + i) * K + (kb + tx)] = (bf16_t)tile[tx][ty + i];
}

// ---------------------------------------------------------------------------
// Small helper kernels
// ---------------------------------------------------------------------------
__global__ void gat_fill(float* __restrict__ p, float v, long long n) {
  long long i = (long long)blockIdx.x * blockDim.x + threadIdx.x;
  long long st = (long long)gridDim.x * blockDim.x;
  for (; i < n; i += st) p[i] = v;
}

__global__ void gat_copy(const float* __restrict__ a, float* __restrict__ b, long long n) {
  long long i = (long long)blockIdx.x * blockDim.x + threadIdx.x;
  long long st = (long long)gridDim.x * blockDim.x;
  for (; i < n; i += st) b[i] = a[i];
}

// out[row] = table[ids[row]]  (512-wide rows)
__global__ void gat_gather_rows(const float* __restrict__ table,
                                const int* __restrict__ ids,
                                float* __restrict__ out, int n_rows) {
  int row = blockIdx.x;
  if (row >= n_rows) return;
  int id = ids[row];
  const float* src = table + (size_t)id * DM;
  float* dst = out + (size_t)row * DM;
  for (int c = threadIdx.x; c < DM; c += blockDim.x) dst[c] = src[c];
}

// score_term[s][h] = sum_d (tab[s] @ feat_fc)[h*64+d] * a_e[h][d]
__global__ void gat_score_term(const float* __restrict__ tab,
                               const float* __restrict__ feat_fc,
                               const float* __restrict__ attn,
                               float* __restrict__ out) {
  int i = threadIdx.x;
  if (i >= 10 * NHEAD) return;
  int s = i >> 3, h = i & 7;
  float acc = 0.0f;
  for (int d = 0; d < HDIM; ++d) {
    float f = 0.0f;
    for (int k = 0; k < 64; ++k)
      f += tab[s * 64 + k] * feat_fc[(size_t)k * DM + h * HDIM + d];
    acc += f * attn[h * 192 + 128 + d];
  }
  out[i] = acc;
}

// zdot[n][h] = sum_d z[n, h*64+d] * attn[h*192 + off + d]
__global__ void gat_head_dot(const float* __restrict__ z,
                             const float* __restrict__ attn, int off,
                             float* __restrict__ out, int n_nodes) {
  int i = blockIdx.x * blockDim.x + threadIdx.x;
  if (i >= n_nodes * NHEAD) return;
  int n = i >> 3, h = i & 7;
  const float* zp = z + (size_t)n * DM + h * HDIM;
  const float* ap = attn + h * 192 + off;
  float s = 0.0f;
  for (int d = 0; d < HDIM; ++d) s += zp[d] * ap[d];
  out[i] = s;
}

__device__ inline void atomicMaxF(float* addr, float val) {
  unsigned int* ua = (unsigned int*)addr;
  unsigned int old = *ua;
  while (__uint_as_float(old) < val) {
    unsigned int assumed = old;
    old = atomicCAS(ua, assumed, __float_as_uint(val));
    if (old == assumed) break;
  }
}

// e = lrelu(zs_dot[src] + score_term[score] (+ zd_dot[dst])); segment max -> m
__global__ void gat_edge_logits(const int* __restrict__ esrc,
                                const int* __restrict__ edst,
                                const int* __restrict__ escore,
                                const float* __restrict__ zs_dot,
                                const float* __restrict__ zd_dot,
                                const float* __restrict__ sterm,
                                float* __restrict__ ebuf,
                                float* __restrict__ mbuf, int E) {
  int i = blockIdx.x * blockDim.x + threadIdx.x;
  if (i >= E * NHEAD) return;
  int e = i >> 3, h = i & 7;
  float v = zs_dot[esrc[e] * NHEAD + h] + sterm[escore[e] * NHEAD + h];
  if (zd_dot) v += zd_dot[edst[e] * NHEAD + h];
  v = (v > 0.0f) ? v : 0.01f * v;
  ebuf[i] = v;
  atomicMaxF(&mbuf[edst[e] * NHEAD + h], v);
}

// ex = exp(e - m[dst]); segment sum -> s  (in place on ebuf)
__global__ void gat_edge_expsum(const int* __restrict__ edst,
                                float* __restrict__ ebuf,
                                const float* __restrict__ mbuf,
                                float* __restrict__ sbuf, int E) {
  int i = blockIdx.x * blockDim.x + threadIdx.x;
  if (i >= E * NHEAD) return;
  int e = i >> 3, h = i & 7;
  float ex = expf(ebuf[i] - mbuf[edst[e] * NHEAD + h]);
  ebuf[i] = ex;
  atomicAdd(&sbuf[edst[e] * NHEAD + h], ex);
}

// hout[dst] += (ex/s[dst]) * z[src]  -- one block per edge
__global__ __launch_bounds__(128) void gat_edge_scatter(
    const int* __restrict__ esrc, const int* __restrict__ edst,
    const float* __restrict__ exbuf, const float* __restrict__ sbuf,
    const float* __restrict__ z, float* __restrict__ hout, int E) {
  int e = blockIdx.x;
  if (e >= E) return;
  int s = esrc[e], t = edst[e];
  const float* zp = z + (size_t)s * DM;
  float* hp = hout + (size_t)t * DM;
  for (int d = threadIdx.x; d < DM; d += blockDim.x) {
    int h = d >> 6;
    float alpha = exbuf[e * NHEAD + h] / sbuf[t * NHEAD + h];
    atomicAdd(&hp[d], alpha * zp[d]);
  }
}

// hres = elu(hout) + tgt
__global__ void gat_elu_res(const float* __restrict__ hout,
                            const float* __restrict__ tgt,
                            float* __restrict__ hres, long long n) {
  long long i = (long long)blockIdx.x * blockDim.x + threadIdx.x;
  long long st = (long long)gridDim.x * blockDim.x;
  for (; i < n; i += st) {
    float x = hout[i];
    x = (x > 0.0f) ? x : expm1f(x);
    hres[i] = x + tgt[i];
  }
}

// out[row] = LayerNorm(o[row] + hres[row]) * g + b
__global__ __launch_bounds__(256) void gat_add_ln(
    const float* __restrict__ o, const float* __restrict__ hres,
    const float* __restrict__ g, const float* __restrict__ b,
    float* __restrict__ out, int n_rows) {
  int row = blockIdx.x;
  if (row >= n_rows) return;
  int t = threadIdx.x;
  __shared__ float red[256];
  size_t base = (size_t)row * DM;
  float x0 = o[base + t]       + hres[base + t];
  float x1 = o[base + t + 256] + hres[base + t + 256];
  red[t] = x0 + x1;
  __syncthreads();
  for (int s = 128; s > 0; s >>= 1) { if (t < s) red[t] += red[t + s]; __syncthreads(); }
  float mu = red[0] * (1.0f / DM);
  __syncthreads();
  float d0 = x0 - mu, d1 = x1 - mu;
  red[t] = d0 * d0 + d1 * d1;
  __syncthreads();
  for (int s = 128; s > 0; s >>= 1) { if (t < s) red[t] += red[t + s]; __syncthreads(); }
  float rs = rsqrtf(red[0] * (1.0f / DM) + 1e-5f);
  out[base + t]       = d0 * rs * g[t]       + b[t];
  out[base + t + 256] = d1 * rs * g[t + 256] + b[t + 256];
}

// ---------------------------------------------------------------------------
// Host-side orchestration
// ---------------------------------------------------------------------------
struct LayerP {
  const float *attn, *b1, *b2, *fc, *feat_fc, *ln_b, *ln_g, *w1, *w2;
};

static LayerP layer_from(void* const* d_in, int base) {
  // jax pytree flattening of a dict is sorted-key order:
  // attn, b1, b2, fc, feat_fc, ln_b, ln_g, w1, w2
  LayerP p;
  p.attn    = (const float*)d_in[base + 0];
  p.b1      = (const float*)d_in[base + 1];
  p.b2      = (const float*)d_in[base + 2];
  p.fc      = (const float*)d_in[base + 3];
  p.feat_fc = (const float*)d_in[base + 4];
  p.ln_b    = (const float*)d_in[base + 5];
  p.ln_g    = (const float*)d_in[base + 6];
  p.w1      = (const float*)d_in[base + 7];
  p.w2      = (const float*)d_in[base + 8];
  return p;
}

static void launch_gemm(const bf16_t* A, const bf16_t* Bt, const float* bias,
                        float* C, int M, int N, int K, int act, hipStream_t s) {
  dim3 grid(cdiv(M, 128), N / 64);
  gat_gemm_bf16<<<grid, 256, 0, s>>>(A, Bt, bias, C, M, N, K, act);
}

static void launch_cvt(const float* in, bf16_t* out, long long n, hipStream_t s) {
  long long blk = (n / 8 + 255) / 256;
  int blocks = (int)(blk > 4096 ? 4096 : blk);
  gat_cvt_bf16<<<blocks, 256, 0, s>>>(in, out, n);
}

static void launch_cvt_t(const float* in, bf16_t* out, int K, int N, hipStream_t s) {
  dim3 grid(K / 32, N / 32);
  gat_cvt_t_bf16<<<grid, 256, 0, s>>>(in, out, K, N);
}

struct Scratch {
  float *z, *hout, *hres, *zs_dot, *zd_dot, *ebuf, *mbuf, *sbuf, *sterm;
  bf16_t *abf1, *abf2, *wtA, *wtB;
  // aliases: obuf = z (free after scatter), tchunk = hout (free after elu_res)
};

static void run_gat_layer(const LayerP& p,
                          const float* src_state, int Ns,
                          const float* tgt_state, int Nt,
                          const int* esrc, const int* edst, const int* escore,
                          int E, const float* tab, bool same_unit,
                          float* out_state, const Scratch& w, hipStream_t s) {
  // z = src @ fc  (Ns x 512), bf16 WMMA path
  launch_cvt(src_state, w.abf1, (long long)Ns * DM, s);
  launch_cvt_t(p.fc, w.wtA, DM, DM, s);
  launch_gemm(w.abf1, w.wtA, nullptr, w.z, Ns, DM, DM, 0, s);
  // per-score x per-head edge-feature term (replaces E x 64 x 512 GEMM)
  gat_score_term<<<1, 128, 0, s>>>(tab, p.feat_fc, p.attn, w.sterm);
  // per-node attention dots
  gat_head_dot<<<cdiv(Ns * NHEAD, 256), 256, 0, s>>>(w.z, p.attn, 0, w.zs_dot, Ns);
  if (same_unit)
    gat_head_dot<<<cdiv(Ns * NHEAD, 256), 256, 0, s>>>(w.z, p.attn, HDIM, w.zd_dot, Ns);
  // segment softmax
  gat_fill<<<512, 256, 0, s>>>(w.mbuf, -1e30f, (long long)Nt * NHEAD);
  gat_fill<<<512, 256, 0, s>>>(w.sbuf, 0.0f,   (long long)Nt * NHEAD);
  gat_fill<<<2048, 256, 0, s>>>(w.hout, 0.0f,  (long long)Nt * DM);
  gat_edge_logits<<<cdiv(E * NHEAD, 256), 256, 0, s>>>(
      esrc, edst, escore, w.zs_dot, same_unit ? w.zd_dot : nullptr,
      w.sterm, w.ebuf, w.mbuf, E);
  gat_edge_expsum<<<cdiv(E * NHEAD, 256), 256, 0, s>>>(edst, w.ebuf, w.mbuf, w.sbuf, E);
  gat_edge_scatter<<<E, 128, 0, s>>>(esrc, edst, w.ebuf, w.sbuf, w.z, w.hout, E);
  // h = elu(h) + tgt
  gat_elu_res<<<2048, 256, 0, s>>>(w.hout, tgt_state, w.hres, (long long)Nt * DM);
  // FFN (chunked): weights converted once, activations per chunk
  float* tchunk = w.hout;  // alias (hout no longer needed)
  float* obuf   = w.z;     // alias (z no longer needed)
  launch_cvt(w.hres, w.abf1, (long long)Nt * DM, s);
  launch_cvt_t(p.w1, w.wtA, DM, FFNH, s);
  launch_cvt_t(p.w2, w.wtB, FFNH, DM, s);
  for (int m0 = 0; m0 < Nt; m0 += FFN_CHUNK) {
    int R = (Nt - m0 < FFN_CHUNK) ? (Nt - m0) : FFN_CHUNK;
    launch_gemm(w.abf1 + (size_t)m0 * DM, w.wtA, p.b1, tchunk, R, FFNH, DM, 1, s);
    launch_cvt(tchunk, w.abf2, (long long)R * FFNH, s);
    launch_gemm(w.abf2, w.wtB, p.b2, obuf + (size_t)m0 * DM, R, DM, FFNH, 0, s);
  }
  // out = LN(o + hres)
  gat_add_ln<<<Nt, 256, 0, s>>>(obuf, w.hres, p.ln_g, p.ln_b, out_state, Nt);
}

extern "C" void kernel_launch(void* const* d_in, const int* in_sizes, int n_in,
                              void* d_out, int out_size, void* d_ws, size_t ws_size,
                              hipStream_t stream) {
  (void)n_in; (void)ws_size;
  const int*   word_ids  = (const int*)d_in[0];
  const float* doc_rep   = (const float*)d_in[1];
  const int*   w2d_src   = (const int*)d_in[2];
  const int*   w2d_dst   = (const int*)d_in[3];
  const int*   w2d_score = (const int*)d_in[4];
  const int*   d2w_src   = (const int*)d_in[5];
  const int*   d2w_dst   = (const int*)d_in[6];
  const int*   d2w_score = (const int*)d_in[7];
  const int*   d2d_src   = (const int*)d_in[8];
  const int*   d2d_dst   = (const int*)d_in[9];
  const int*   d2d_score = (const int*)d_in[10];
  // params (sorted-key pytree): d2d layer, d2d_tab, d2w layer, w2d layer,
  // w2d_tab, word_embed
  LayerP p_d2d = layer_from(d_in, 11);
  const float* d2d_tab = (const float*)d_in[20];
  LayerP p_d2w = layer_from(d_in, 21);
  LayerP p_w2d = layer_from(d_in, 30);
  const float* w2d_tab = (const float*)d_in[39];
  const float* word_embed = (const float*)d_in[40];

  const int NW = in_sizes[0];           // 20000
  const int ND = in_sizes[1] / DM;      // 1024
  const int E_wd = in_sizes[2];         // 131072
  const int E_dd = in_sizes[8];         // 32768

  // workspace bump allocator (16-byte aligned chunks)
  float* base = (float*)d_ws;
  size_t off = 0;
  auto alloc = [&](size_t n) {
    off = (off + 3) & ~(size_t)3;
    float* p = base + off; off += n; return p;
  };
  float* wstateA = alloc((size_t)NW * DM);
  float* wstateB = alloc((size_t)NW * DM);
  float* dstateA = alloc((size_t)ND * DM);
  float* dstateB = alloc((size_t)ND * DM);
  Scratch w;
  w.z      = alloc((size_t)NW * DM);
  w.hout   = alloc((size_t)NW * DM);          // also tchunk (CH*2048 <= NW*512)
  w.hres   = alloc((size_t)NW * DM);
  w.zs_dot = alloc((size_t)NW * NHEAD);
  w.zd_dot = alloc((size_t)NW * NHEAD);
  w.ebuf   = alloc((size_t)E_wd * NHEAD);
  w.mbuf   = alloc((size_t)NW * NHEAD);
  w.sbuf   = alloc((size_t)NW * NHEAD);
  w.sterm  = alloc(128);
  w.abf1   = (bf16_t*)alloc((size_t)NW * DM / 2);          // NW*512 bf16
  w.abf2   = (bf16_t*)alloc((size_t)FFN_CHUNK * FFNH / 2); // CH*2048 bf16
  w.wtA    = (bf16_t*)alloc((size_t)DM * FFNH / 2);        // 512*2048 bf16
  w.wtB    = (bf16_t*)alloc((size_t)FFNH * DM / 2);        // 2048*512 bf16

  // initial states
  gat_gather_rows<<<NW, 256, 0, stream>>>(word_embed, word_ids, wstateA, NW);
  gat_copy<<<1024, 256, 0, stream>>>(doc_rep, dstateA, (long long)ND * DM);

  float* wcur = wstateA; float* wnxt = wstateB;
  float* dcur = dstateA; float* dnxt = dstateB;

  for (int it = 0; it < 2; ++it) {
    // word_state = d2w(src=doc, tgt=word), edge table = w2d_tab
    run_gat_layer(p_d2w, dcur, ND, wcur, NW, d2w_src, d2w_dst, d2w_score,
                  E_wd, w2d_tab, false, wnxt, w, stream);
    { float* t = wcur; wcur = wnxt; wnxt = t; }
    // doc_state = w2d(src=word, tgt=doc)
    run_gat_layer(p_w2d, wcur, NW, dcur, ND, w2d_src, w2d_dst, w2d_score,
                  E_wd, w2d_tab, false, dnxt, w, stream);
    { float* t = dcur; dcur = dnxt; dnxt = t; }
    // doc_state = d2d(src=tgt=doc, same_unit)
    run_gat_layer(p_d2d, dcur, ND, dcur, ND, d2d_src, d2d_dst, d2d_score,
                  E_dd, d2d_tab, true, dnxt, w, stream);
    { float* t = dcur; dcur = dnxt; dnxt = t; }
  }

  gat_copy<<<1024, 256, 0, stream>>>(dcur, (float*)d_out, (long long)out_size);
}
